// GNNEncoder_32762010534265
// MI455X (gfx1250) — matrix operand order repaired and checked by reading, not compiled, not run on hardware
//
#include <hip/hip_runtime.h>

typedef __attribute__((ext_vector_type(16))) __bf16 v16bf;
typedef __attribute__((ext_vector_type(8)))  float  v8f;

#define FD 128   // feature dim

// ---------------------------------------------------------------------------
// Utility kernels
// ---------------------------------------------------------------------------
__global__ __launch_bounds__(256) void zero_f32(float* __restrict__ p, int n) {
    int i = blockIdx.x * 256 + threadIdx.x;
    if (i < n) p[i] = 0.0f;
}

__global__ __launch_bounds__(256) void degree_kernel(const int* __restrict__ col,
                                                     float* __restrict__ deg, int nE) {
    int i = blockIdx.x * 256 + threadIdx.x;
    if (i < nE) unsafeAtomicAdd(&deg[col[i]], 1.0f);
}

__global__ __launch_bounds__(256) void dinv_kernel(float* __restrict__ deg, int nN) {
    int i = blockIdx.x * 256 + threadIdx.x;
    if (i < nN) {
        float d = deg[i];
        deg[i] = (d > 0.0f) ? __frsqrt_rn(d) : 0.0f;   // in-place deg -> deg^-1/2
    }
}

__global__ __launch_bounds__(256) void norm_kernel(const int* __restrict__ row,
                                                   const int* __restrict__ col,
                                                   const float* __restrict__ dinv,
                                                   float* __restrict__ nw, int nE) {
    int i = blockIdx.x * 256 + threadIdx.x;
    if (i < nE) nw[i] = dinv[row[i]] * dinv[col[i]];
}

// ---------------------------------------------------------------------------
// Weight conversion fp32 -> bf16, pre-swizzled into the WMMA B-operand layout.
// B (32x16 K x N, 16-bit), this code's k order: lanes 0-15 hold K=0..15, lanes 16-31 hold K=16..31
// of each 32-K tile; N = nt*16 + (lane&15). Stored so each lane reads its 16
// halves (32B) contiguously:  dst[(((kt*8+nt)*32 + lane)*16 + h]
// ---------------------------------------------------------------------------
__global__ __launch_bounds__(256) void convert_weights(const float* __restrict__ W,
                                                       __bf16* __restrict__ Wb, int nMat) {
    int tid = blockIdx.x * 256 + threadIdx.x;
    int total = nMat * FD * FD;
    if (tid >= total) return;
    int h    = tid & 15;
    int lane = (tid >> 4) & 31;
    int nt   = (tid >> 9) & 7;
    int kt   = (tid >> 12) & 3;
    int m    = tid >> 14;
    int K = kt * 32 + ((lane >> 4) << 4) + h;
    int N = nt * 16 + (lane & 15);
    Wb[tid] = (__bf16)W[(size_t)m * (FD * FD) + (size_t)K * FD + N];
}

// ---------------------------------------------------------------------------
// Edge propagation: out[col] += norm_w * cur[row].  One wave per edge,
// one float4 per lane (32 lanes x 4 = 128 features), hw f32 atomics.
// Working set fits in the 192MB L2 -> bound by L2 atomic throughput.
// ---------------------------------------------------------------------------
__global__ __launch_bounds__(256) void propagate_kernel(const int* __restrict__ row,
                                                        const int* __restrict__ col,
                                                        const float* __restrict__ nw,
                                                        const float* __restrict__ cur,
                                                        float* __restrict__ out, int nE) {
    int e = blockIdx.x * 8 + (threadIdx.x >> 5);
    if (e >= nE) return;
    int lane = threadIdx.x & 31;
    int r = row[e];
    int c = col[e];
    float w = nw[e];
    float4 v = ((const float4*)(cur + (size_t)r * FD))[lane];
    float* dst = out + (size_t)c * FD + lane * 4;
    unsafeAtomicAdd(dst + 0, w * v.x);
    unsafeAtomicAdd(dst + 1, w * v.y);
    unsafeAtomicAdd(dst + 2, w * v.z);
    unsafeAtomicAdd(dst + 3, w * v.w);
}

// ---------------------------------------------------------------------------
// WMMA GEMM: Y[16-row strip] (+)= X @ W  (X fp32 -> bf16 on the fly, W
// pre-swizzled bf16, fp32 accumulate).  One wave owns a 16x128 strip:
// v8f acc[8], K-loop of 4 x v_wmma_f32_16x16x32_bf16 per column tile.
// Optional bias (broadcast into C) and fused ReLU epilogue.
//
// A k order used here, same as B (16-bit 16x32): lane&15 = M; per-lane halves h=0..15 map to
//   K = (lane>>4)*16 + h   (pairs K-contiguous)
// C/D layout (f32 16x16): N = lane&15, M = (lane>>4)*8 + vgpr.
// ---------------------------------------------------------------------------
__global__ __launch_bounds__(256) void tag_gemm128(const float* __restrict__ X,
                                                   const __bf16* __restrict__ Wb,
                                                   const float* __restrict__ bias,
                                                   float* __restrict__ Y,
                                                   int nRows, int accumulate, int relu) {
    const int wave   = threadIdx.x >> 5;
    const int lane   = threadIdx.x & 31;
    const int tile   = blockIdx.x * 8 + wave;
    if (tile * 16 >= nRows) return;                 // wave-uniform: EXEC stays full
    const int m0     = tile * 16;
    const int laneHi = lane >> 4;
    const int mr     = lane & 15;

    v8f acc[8];
    if (accumulate) {
        #pragma unroll
        for (int nt = 0; nt < 8; ++nt)
            #pragma unroll
            for (int r = 0; r < 8; ++r)
                acc[nt][r] = Y[(size_t)(m0 + laneHi * 8 + r) * FD + nt * 16 + mr];
    } else if (bias) {
        #pragma unroll
        for (int nt = 0; nt < 8; ++nt) {
            float bv = bias[nt * 16 + mr];
            #pragma unroll
            for (int r = 0; r < 8; ++r) acc[nt][r] = bv;
        }
    } else {
        #pragma unroll
        for (int nt = 0; nt < 8; ++nt)
            #pragma unroll
            for (int r = 0; r < 8; ++r) acc[nt][r] = 0.0f;
    }

    const float* xrow = X + (size_t)(m0 + mr) * FD;

    #pragma unroll
    for (int kt = 0; kt < 4; ++kt) {
        // Build bf16 A fragment from fp32 activations (pairs are K-contiguous).
        v16bf a;
        #pragma unroll
        for (int p = 0; p < 8; ++p) {
            int k = kt * 32 + (laneHi << 4) + (p << 1);
            float2 f = *(const float2*)(xrow + k);
            a[2 * p]     = (__bf16)f.x;
            a[2 * p + 1] = (__bf16)f.y;
        }
        #pragma unroll
        for (int nt = 0; nt < 8; ++nt) {
            const v16bf b = *(const v16bf*)(Wb + (((size_t)(kt * 8 + nt) * 32 + lane) << 4));
            acc[nt] = __builtin_amdgcn_wmma_f32_16x16x32_bf16(
                false, a, false, b, (short)0, acc[nt], false, false);
        }
    }

    #pragma unroll
    for (int nt = 0; nt < 8; ++nt)
        #pragma unroll
        for (int r = 0; r < 8; ++r) {
            float v = acc[nt][r];
            if (relu) v = v > 0.0f ? v : 0.0f;
            Y[(size_t)(m0 + laneHi * 8 + r) * FD + nt * 16 + mr] = v;
        }
}

// ---------------------------------------------------------------------------
// Host orchestration
// ---------------------------------------------------------------------------
extern "C" void kernel_launch(void* const* d_in, const int* in_sizes, int n_in,
                              void* d_out, int out_size, void* d_ws, size_t ws_size,
                              hipStream_t stream) {
    const float* x  = (const float*)d_in[0];
    const int*   ei = (const int*)d_in[1];
    const float* W1 = (const float*)d_in[2];
    const float* b1 = (const float*)d_in[3];
    const float* W2 = (const float*)d_in[4];
    const float* b2 = (const float*)d_in[5];
    const float* W3 = (const float*)d_in[6];
    const float* b3 = (const float*)d_in[7];
    const float* Wf = (const float*)d_in[8];
    const float* bf = (const float*)d_in[9];

    const int nN = in_sizes[0] / FD;     // 50000
    const int nE = in_sizes[1] / 2;      // 1600000
    const int* row = ei;
    const int* col = ei + nE;

    // Workspace carve-out (256B aligned)
    char*  ws  = (char*)d_ws;
    size_t off = 0;
    auto carve = [&](size_t bytes) -> void* {
        void* p = ws + off;
        off = (off + bytes + 255) & ~(size_t)255;
        return p;
    };
    const size_t matHalves = (size_t)FD * FD;                 // 16384 halves/matrix
    __bf16* Wb   = (__bf16*)carve(13 * matHalves * sizeof(__bf16));
    float*  dinv = (float*)carve((size_t)nN * sizeof(float));
    float*  nw   = (float*)carve((size_t)nE * sizeof(float));
    float*  P0   = (float*)carve((size_t)nN * FD * sizeof(float));
    float*  P1   = (float*)carve((size_t)nN * FD * sizeof(float));
    float*  A1   = (float*)carve((size_t)nN * FD * sizeof(float));
    float*  A2   = (float*)carve((size_t)nN * FD * sizeof(float));
    (void)ws_size; (void)n_in; (void)out_size;

    const int gbN  = (nN + 255) / 256;
    const int gbE  = (nE + 255) / 256;
    const int gbNF = (nN * FD + 255) / 256;
    const int tiles = (nN + 15) / 16;
    const int gbG  = (tiles + 7) / 8;
    const int gbP  = (nE + 7) / 8;

    // 1) Weights fp32 -> bf16, swizzled to the WMMA B-operand layout
    convert_weights<<<(4 * FD * FD + 255) / 256, 256, 0, stream>>>(W1, Wb + 0 * matHalves, 4);
    convert_weights<<<(4 * FD * FD + 255) / 256, 256, 0, stream>>>(W2, Wb + 4 * matHalves, 4);
    convert_weights<<<(4 * FD * FD + 255) / 256, 256, 0, stream>>>(W3, Wb + 8 * matHalves, 4);
    convert_weights<<<(1 * FD * FD + 255) / 256, 256, 0, stream>>>(Wf, Wb + 12 * matHalves, 1);

    // 2) gcn_norm: deg -> deg^-1/2 -> norm_w
    zero_f32<<<gbN, 256, 0, stream>>>(dinv, nN);
    degree_kernel<<<gbE, 256, 0, stream>>>(col, dinv, nE);
    dinv_kernel<<<gbN, 256, 0, stream>>>(dinv, nN);
    norm_kernel<<<gbE, 256, 0, stream>>>(row, col, dinv, nw, nE);

    // 3) One TAGConv layer: out = relu( sum_k (A^k hin) @ W[k] + b )
    auto run_layer = [&](const float* hin, const __bf16* Wm, const float* bias,
                         float* out) {
        tag_gemm128<<<gbG, 256, 0, stream>>>(hin, Wm, bias, out, nN, 0, 0);
        const float* cur = hin;
        float* tgt[3] = {P0, P1, P0};
        for (int k = 1; k <= 3; ++k) {
            float* t = tgt[k - 1];
            zero_f32<<<gbNF, 256, 0, stream>>>(t, nN * FD);
            propagate_kernel<<<gbP, 256, 0, stream>>>(row, col, nw, cur, t, nE);
            tag_gemm128<<<gbG, 256, 0, stream>>>(t, Wm + (size_t)k * matHalves,
                                                 nullptr, out, nN, 1, (k == 3) ? 1 : 0);
            cur = t;
        }
    };

    run_layer(x,  Wb + 0 * matHalves, b1, A1);
    run_layer(A1, Wb + 4 * matHalves, b2, A2);
    run_layer(A2, Wb + 8 * matHalves, b3, A1);

    // 4) Final linear: z = h @ Wf + bf   (no relu) -> d_out
    tag_gemm128<<<gbG, 256, 0, stream>>>(A1, Wb + 12 * matHalves, bf,
                                         (float*)d_out, nN, 0, 0);
}
